// SpectralConv2d_87213605913221
// MI455X (gfx1250) — matrix operand order repaired and checked by reading, not compiled, not run on hardware
//
#include <hip/hip_runtime.h>
#include <hip/hip_bf16.h>

// ---------------------------------------------------------------------------
// FNO SpectralConv2d block for MI455X (gfx1250, wave32, WMMA).
// B=32, M=N=128, I=O=64, NM=16 modes.
//
// Pipeline (all on `stream`, in order):
//   k1  res = x @ w_lin + b_lin                (WMMA f16->f32)   -> d_out[0:]
//   k2  truncated DFT along N (kx=0..15)       (VALU f32, LDS)   -> ws
//   k3  truncated DFT along M (32 ky modes)    (VALU f32, LDS)   -> ws
//   k4a factorized complex mix, stage 1 (w0)   (VALU f32, LDS)   -> ws
//   k4b factorized complex mix, stage 2 (w1)   (VALU f32, LDS)   -> ws
//   k5  inverse DFT along ky (full 128 u)      (VALU f32, LDS)   -> ws
//   k6  inverse rDFT along kx + res + relu     (VALU f32, LDS)   -> d_out[half:]
//   k7  two MLPs 64->128->64                   (WMMA f16->f32)   -> d_out
//
// d_out aliasing: res lives in the bcast half, y_act in the fcast half.
// k7 block i reads exactly rows [128i,128i+128) of y_act before writing the
// same rows of fcast, and res is dead once k7 writes bcast => race-free.
// ---------------------------------------------------------------------------

typedef __attribute__((ext_vector_type(16))) _Float16 v16h;
typedef __attribute__((ext_vector_type(8)))  float    v8f;

#define TWOPI_128 0.04908738521234052f  // 2*pi/128

// --- WMMA fragment loaders (ISA 7.12.2 16-bit layouts) ---------------------
__device__ __forceinline__ int kmapA(int h, int j) {
  return ((j & 8) << 1) + (h << 3) + (j & 7);   // A 16x32: K per half j
}
__device__ __forceinline__ int kmapB(int h, int j) {
  return ((j & 8) << 1) + (h << 3) + (j & 7);   // B 32x16: K per half j
}
__device__ __forceinline__ v16h ldA_f32(const float* __restrict__ p, int stride, int h) {
  v16h f;
#pragma unroll
  for (int j = 0; j < 16; ++j) f[j] = (_Float16)p[kmapA(h, j) * stride];
  return f;
}
__device__ __forceinline__ v16h ldB_f32(const float* __restrict__ p, int stride, int h) {
  v16h f;
#pragma unroll
  for (int j = 0; j < 16; ++j) f[j] = (_Float16)p[kmapB(h, j) * stride];
  return f;
}
__device__ __forceinline__ v16h ldA_h16(const _Float16* __restrict__ p, int h) {
  v16h f;
#pragma unroll
  for (int j = 0; j < 16; ++j) f[j] = p[kmapA(h, j)];
  return f;
}
#define WMMA_F32F16(a, b, c) \
  __builtin_amdgcn_wmma_f32_16x16x32_f16(false, (a), false, (b), (short)0, (c), false, false)

// ---------------------------------------------------------------------------
// k1: res[row, o] = x[row, :] @ w_lin + b_lin   (rows = B*M*N = 524288)
// 8 waves/block, 16 rows/wave, 128 rows/block, grid = 4096.
// ---------------------------------------------------------------------------
__global__ __launch_bounds__(256) void k1_res(const float* __restrict__ x,
                                              const float* __restrict__ wlin,
                                              const float* __restrict__ blin,
                                              float* __restrict__ res) {
  int wave = threadIdx.x >> 5, lane = threadIdx.x & 31;
  int h = lane >> 4, ln = lane & 15;
  long r0 = ((long)blockIdx.x * 8 + wave) * 16;
  const float* arow = x + (r0 + ln) * 64;
  v16h a0 = ldA_f32(arow,      1, h);   // K = 0..31
  v16h a1 = ldA_f32(arow + 32, 1, h);   // K = 32..63
#pragma unroll
  for (int nt = 0; nt < 4; ++nt) {
    v8f c = {};
    v16h b0 = ldB_f32(wlin +            nt * 16 + ln, 64, h);
    v16h b1 = ldB_f32(wlin + 32 * 64 + nt * 16 + ln, 64, h);
    c = WMMA_F32F16(a0, b0, c);
    c = WMMA_F32F16(a1, b1, c);
    float bias = blin[nt * 16 + ln];
#pragma unroll
    for (int d = 0; d < 8; ++d)
      res[(r0 + d + h * 8) * 64 + nt * 16 + ln] = c[d] + bias;
  }
}

// ---------------------------------------------------------------------------
// k2: truncated DFT over n (rfft axis): XA[b,m,kx,i] = sum_n x[b,m,n,i] e^{-2pi i kx n/128}
// block per (b,m); x slice staged in LDS (32 KB). grid = 4096.
// ---------------------------------------------------------------------------
__global__ __launch_bounds__(256) void k2_dftn(const float* __restrict__ x,
                                               float* __restrict__ XAre,
                                               float* __restrict__ XAim) {
  __shared__ float xs[128 * 64];
  int t = threadIdx.x;
  long bm = blockIdx.x;
  const float* src = x + bm * (128 * 64);
  for (int p = t; p < 128 * 64; p += 256) xs[p] = src[p];
  __syncthreads();
  int k = t >> 4;            // kx 0..15
  int i0 = (t & 15) * 4;     // 4 consecutive channels
  float ar[4] = {0.f, 0.f, 0.f, 0.f}, ai[4] = {0.f, 0.f, 0.f, 0.f};
  float step = -TWOPI_128 * (float)k;   // e^{-i theta}
  for (int n = 0; n < 128; ++n) {
    float s, c;
    __sincosf(step * (float)n, &s, &c);
    const float* row = &xs[n * 64 + i0];
#pragma unroll
    for (int q = 0; q < 4; ++q) { ar[q] += row[q] * c; ai[q] += row[q] * s; }
  }
  long ob = (bm * 16 + k) * 64 + i0;
#pragma unroll
  for (int q = 0; q < 4; ++q) { XAre[ob + q] = ar[q]; XAim[ob + q] = ai[q]; }
}

// ---------------------------------------------------------------------------
// k3: truncated DFT over m: XF[b,r,ky,kx,i] = (1/128) sum_m XA[b,m,kx,i] e^{-2pi i ky m/128}
// ky in {0..15} (r=0) or {112..127} == {-16..-1} (r=1). block per (b,kx), grid 512.
// XA m-slices staged through LDS in two 64-row chunks (32 KB).
// ---------------------------------------------------------------------------
__global__ __launch_bounds__(256) void k3_dftm(const float* __restrict__ XAre,
                                               const float* __restrict__ XAim,
                                               float* __restrict__ XFre,
                                               float* __restrict__ XFim) {
  __shared__ float xr[64 * 64], xi[64 * 64];
  int t = threadIdx.x;
  int b = blockIdx.x >> 4;
  int kx = blockIdx.x & 15;
  int rk = t >> 3;                 // 0..31 = (r,ky)
  int r = rk >> 4, k = rk & 15;
  int i0 = (t & 7) * 8;            // 8 channels
  float kyeff = r ? (float)(k - 16) : (float)k;
  float step = -TWOPI_128 * kyeff;
  float ar[8], ai[8];
#pragma unroll
  for (int q = 0; q < 8; ++q) { ar[q] = 0.f; ai[q] = 0.f; }
  for (int ch = 0; ch < 2; ++ch) {
    __syncthreads();
    for (int p = t; p < 64 * 64; p += 256) {
      int mm = p >> 6, i = p & 63;
      long idx = (((long)b * 128 + ch * 64 + mm) * 16 + kx) * 64 + i;
      xr[p] = XAre[idx];
      xi[p] = XAim[idx];
    }
    __syncthreads();
    for (int mm = 0; mm < 64; ++mm) {
      float s, c;
      __sincosf(step * (float)(ch * 64 + mm), &s, &c);
      const float* pr = &xr[mm * 64 + i0];
      const float* pi = &xi[mm * 64 + i0];
#pragma unroll
      for (int q = 0; q < 8; ++q) {
        ar[q] += pr[q] * c - pi[q] * s;  // (xr+i*xi)*(c+i*s), s has e^{-i} sign folded
        ai[q] += pi[q] * c + pr[q] * s;
      }
    }
  }
  const float scale = 1.0f / 128.0f;   // ortho norm for both forward axes
  long ob = ((((long)b * 2 + r) * 16 + k) * 16 + kx) * 64 + i0;
#pragma unroll
  for (int q = 0; q < 8; ++q) { XFre[ob + q] = ar[q] * scale; XFim[ob + q] = ai[q] * scale; }
}

// ---------------------------------------------------------------------------
// k4a: stage-1 mixing: C1[b,r,kx,o,ky] = sum_i w0[i,o,ky] * XF[b,r,ky,kx,i]  (complex)
// block per (r,ky,b): weights (32KB) + activation slice (8KB) in LDS. grid 1024.
// ---------------------------------------------------------------------------
__global__ __launch_bounds__(256) void k4_stage1(const float* __restrict__ XFre,
                                                 const float* __restrict__ XFim,
                                                 const float* __restrict__ ws0,
                                                 const float* __restrict__ ws1,
                                                 float* __restrict__ C1re,
                                                 float* __restrict__ C1im) {
  __shared__ float wr[64 * 64], wi[64 * 64], ar[16 * 64], ai[16 * 64];
  int t = threadIdx.x;
  int b = blockIdx.x & 31;
  int ky = (blockIdx.x >> 5) & 15;
  int r = blockIdx.x >> 9;
  const float* wsp = r ? ws1 : ws0;            // w_spec[r][0][i][o][ky][c]
  for (int p = t; p < 4096; p += 256) {
    int i = p >> 6, o = p & 63;
    long widx = (((long)i * 64 + o) * 16 + ky) * 2;   // factor 0
    wr[p] = wsp[widx];
    wi[p] = wsp[widx + 1];
  }
  long ab = ((((long)b * 2 + r) * 16 + ky) * 16) * 64;
  for (int p = t; p < 1024; p += 256) { ar[p] = XFre[ab + p]; ai[p] = XFim[ab + p]; }
  __syncthreads();
  int o = t & 63, kxg = t >> 6;
#pragma unroll
  for (int q = 0; q < 4; ++q) {
    int kx = kxg * 4 + q;
    float cr = 0.f, ci = 0.f;
    for (int i = 0; i < 64; ++i) {
      float wre = wr[i * 64 + o], wim = wi[i * 64 + o];
      float are = ar[kx * 64 + i], aim = ai[kx * 64 + i];
      cr += wre * are - wim * aim;
      ci += wre * aim + wim * are;
    }
    long oidx = ((((long)b * 2 + r) * 16 + kx) * 64 + o) * 16 + ky;
    C1re[oidx] = cr;
    C1im[oidx] = ci;
  }
}

// ---------------------------------------------------------------------------
// k4b: stage-2 mixing: G[b,kx,r,ky,o] = sum_c w1[c,o,kx] * C1[b,r,kx,c,ky]  (complex)
// block per (r,kx,b). grid 1024.
// ---------------------------------------------------------------------------
__global__ __launch_bounds__(256) void k4_stage2(const float* __restrict__ C1re,
                                                 const float* __restrict__ C1im,
                                                 const float* __restrict__ ws0,
                                                 const float* __restrict__ ws1,
                                                 float* __restrict__ Gre,
                                                 float* __restrict__ Gim) {
  __shared__ float wr[64 * 64], wi[64 * 64], ar[64 * 16], ai[64 * 16];
  int t = threadIdx.x;
  int b = blockIdx.x & 31;
  int kx = (blockIdx.x >> 5) & 15;
  int r = blockIdx.x >> 9;
  const float* wsp = r ? ws1 : ws0;            // w_spec[r][1][i][o][kx][c]
  for (int p = t; p < 4096; p += 256) {
    int i = p >> 6, o = p & 63;
    long widx = ((((long)64 + i) * 64 + o) * 16 + kx) * 2;   // factor 1
    wr[p] = wsp[widx];
    wi[p] = wsp[widx + 1];
  }
  long ab = (((long)b * 2 + r) * 16 + kx) * 1024;            // [c*16+ky]
  for (int p = t; p < 1024; p += 256) { ar[p] = C1re[ab + p]; ai[p] = C1im[ab + p]; }
  __syncthreads();
  int o = t & 63, kyg = t >> 6;
#pragma unroll
  for (int q = 0; q < 4; ++q) {
    int ky = kyg * 4 + q;
    float cr = 0.f, ci = 0.f;
    for (int c = 0; c < 64; ++c) {
      float wre = wr[c * 64 + o], wim = wi[c * 64 + o];
      float are = ar[c * 16 + ky], aim = ai[c * 16 + ky];
      cr += wre * are - wim * aim;
      ci += wre * aim + wim * are;
    }
    long oidx = ((((long)b * 16 + kx) * 2 + r) * 16 + ky) * 64 + o;
    Gre[oidx] = cr;
    Gim[oidx] = ci;
  }
}

// ---------------------------------------------------------------------------
// k5: inverse DFT over ky: Y1[b,u,kx,o] = sum_{32 modes} G * e^{+2pi i ky u/128}
// block per (b,kx); G slice (16KB) in LDS. grid 512.
// ---------------------------------------------------------------------------
__global__ __launch_bounds__(256) void k5_iffty(const float* __restrict__ Gre,
                                                const float* __restrict__ Gim,
                                                float* __restrict__ Y1re,
                                                float* __restrict__ Y1im) {
  __shared__ float gr[2048], gi[2048];          // [(r*16+ky)*64+o]
  int t = threadIdx.x;
  int b = blockIdx.x >> 4;
  int kx = blockIdx.x & 15;
  long gb = ((long)b * 16 + kx) * 2048;
  for (int p = t; p < 2048; p += 256) { gr[p] = Gre[gb + p]; gi[p] = Gim[gb + p]; }
  __syncthreads();
  int u = t & 127, og = t >> 7;
  float cs[32], sn[32];
#pragma unroll
  for (int j = 0; j < 32; ++j) {
    int r = j >> 4, k = j & 15;
    float kyeff = r ? (float)(k - 16) : (float)k;
    __sincosf(TWOPI_128 * kyeff * (float)u, &sn[j], &cs[j]);
  }
  for (int oo = 0; oo < 32; ++oo) {
    int o = og * 32 + oo;
    float yr = 0.f, yi = 0.f;
#pragma unroll
    for (int j = 0; j < 32; ++j) {
      float grv = gr[j * 64 + o], giv = gi[j * 64 + o];
      yr += grv * cs[j] - giv * sn[j];          // e^{+i theta}
      yi += grv * sn[j] + giv * cs[j];
    }
    long oidx = (((long)b * 128 + u) * 16 + kx) * 64 + o;
    Y1re[oidx] = yr;
    Y1im[oidx] = yi;
  }
}

// ---------------------------------------------------------------------------
// k6: inverse rDFT over kx (Hermitian) + residual + relu:
//   y[b,u,v,o] = relu( res + (1/128)*( Y1[0] + 2*sum_{k=1..15} Re{Y1[k] e^{+2pi i k v/128}} ) )
// block per (b,u); Y1 slice (8KB) in LDS. grid 4096.
// ---------------------------------------------------------------------------
__global__ __launch_bounds__(256) void k6_irfftx(const float* __restrict__ Y1re,
                                                 const float* __restrict__ Y1im,
                                                 const float* __restrict__ res,
                                                 float* __restrict__ yact) {
  __shared__ float lr[1024], li[1024];          // [kx*64+o]
  int t = threadIdx.x;
  int b = blockIdx.x >> 7;
  int u = blockIdx.x & 127;
  long yb = ((long)b * 128 + u) * 1024;
  for (int p = t; p < 1024; p += 256) { lr[p] = Y1re[yb + p]; li[p] = Y1im[yb + p]; }
  __syncthreads();
  int v = t & 127, og = t >> 7;
  float cs[15], sn[15];
#pragma unroll
  for (int k = 1; k < 16; ++k)
    __sincosf(TWOPI_128 * (float)k * (float)v, &sn[k - 1], &cs[k - 1]);
  const float scale = 1.0f / 128.0f;            // ortho norm for both inverse axes
  for (int oo = 0; oo < 32; ++oo) {
    int o = og * 32 + oo;
    float val = lr[o];                          // k=0 term
#pragma unroll
    for (int k = 1; k < 16; ++k)
      val += 2.f * (lr[k * 64 + o] * cs[k - 1] - li[k * 64 + o] * sn[k - 1]);
    long idx = ((((long)b * 128 + u) * 128) + v) * 64 + o;
    float rv = res[idx] + val * scale;
    yact[idx] = rv > 0.f ? rv : 0.f;
  }
}

// ---------------------------------------------------------------------------
// k7: two fused MLPs (64 -> relu(128) -> 64) over 524288 rows, WMMA f16.
// Hidden tile round-trips through LDS to re-layout C (v8f) into A fragments.
// 8 waves/block, 16 rows/wave. grid 4096. 64 v_wmma per wave.
// ---------------------------------------------------------------------------
__global__ __launch_bounds__(256) void k7_mlp(const float* __restrict__ yact,
                                              const float* __restrict__ w1b, const float* __restrict__ b1b,
                                              const float* __restrict__ w2b, const float* __restrict__ b2b,
                                              const float* __restrict__ w1f, const float* __restrict__ b1f,
                                              const float* __restrict__ w2f, const float* __restrict__ b2f,
                                              float* __restrict__ outb, float* __restrict__ outf) {
  __shared__ _Float16 hbuf[8][16][128];         // per-wave 16x128 hidden tile (32 KB)
  int wave = threadIdx.x >> 5, lane = threadIdx.x & 31;
  int h = lane >> 4, ln = lane & 15;
  long r0 = ((long)blockIdx.x * 8 + wave) * 16;
  const float* arow = yact + (r0 + ln) * 64;
  v16h a0 = ldA_f32(arow,      1, h);
  v16h a1 = ldA_f32(arow + 32, 1, h);
  for (int br = 0; br < 2; ++br) {
    const float* W1 = br ? w1f : w1b;
    const float* B1 = br ? b1f : b1b;
    const float* W2 = br ? w2f : w2b;
    const float* B2 = br ? b2f : b2b;
    float* out = br ? outf : outb;
    // hidden = relu(y @ W1 + B1), 8 N-tiles of 16
#pragma unroll
    for (int nt = 0; nt < 8; ++nt) {
      v8f c = {};
      v16h bb0 = ldB_f32(W1 +            nt * 16 + ln, 128, h);
      v16h bb1 = ldB_f32(W1 + 32 * 128 + nt * 16 + ln, 128, h);
      c = WMMA_F32F16(a0, bb0, c);
      c = WMMA_F32F16(a1, bb1, c);
      float bias = B1[nt * 16 + ln];
#pragma unroll
      for (int d = 0; d < 8; ++d) {
        float v = c[d] + bias;
        hbuf[wave][d + h * 8][nt * 16 + ln] = (_Float16)(v > 0.f ? v : 0.f);
      }
    }
    __syncthreads();
    v16h a2[4];
#pragma unroll
    for (int kc = 0; kc < 4; ++kc)
      a2[kc] = ldA_h16(&hbuf[wave][ln][kc * 32], h);
    __syncthreads();
    // out = hidden @ W2 + B2, 4 N-tiles of 16, K = 128 (4 chunks)
#pragma unroll
    for (int nt = 0; nt < 4; ++nt) {
      v8f c = {};
#pragma unroll
      for (int kc = 0; kc < 4; ++kc) {
        v16h bb = ldB_f32(W2 + kc * 32 * 64 + nt * 16 + ln, 64, h);
        c = WMMA_F32F16(a2[kc], bb, c);
      }
      float bias = B2[nt * 16 + ln];
#pragma unroll
      for (int d = 0; d < 8; ++d)
        out[(r0 + d + h * 8) * 64 + nt * 16 + ln] = c[d] + bias;
    }
  }
}

// ---------------------------------------------------------------------------
extern "C" void kernel_launch(void* const* d_in, const int* in_sizes, int n_in,
                              void* d_out, int out_size, void* d_ws, size_t ws_size,
                              hipStream_t stream) {
  const float* x    = (const float*)d_in[0];
  const float* ws0  = (const float*)d_in[1];
  const float* ws1  = (const float*)d_in[2];
  const float* wlin = (const float*)d_in[3];
  const float* blin = (const float*)d_in[4];
  const float* wb1  = (const float*)d_in[5];
  const float* bb1  = (const float*)d_in[6];
  const float* wb2  = (const float*)d_in[7];
  const float* bb2  = (const float*)d_in[8];
  const float* wf1  = (const float*)d_in[9];
  const float* bf1  = (const float*)d_in[10];
  const float* wf2  = (const float*)d_in[11];
  const float* bf2  = (const float*)d_in[12];

  float* out  = (float*)d_out;
  float* res  = out;               // bcast half doubles as res scratch (dead after k7 writes it)
  float* yact = out + 33554432;    // fcast half doubles as y_act scratch (per-block-row safe)

  // workspace: 23,068,672 floats = 88 MB
  float* w    = (float*)d_ws;
  float* XAre = w;                    // 4,194,304  [B,M,16,I]
  float* XAim = XAre + 4194304;
  float* XFre = XAim + 4194304;       // 1,048,576  [B,2,16ky,16kx,I]
  float* XFim = XFre + 1048576;
  float* C1re = XFim + 1048576;       // 1,048,576  [B,2,16kx,64,16ky]
  float* C1im = C1re + 1048576;
  float* Gre  = C1im + 1048576;       // 1,048,576  [B,16kx,2,16ky,64]
  float* Gim  = Gre  + 1048576;
  float* Y1re = Gim  + 1048576;       // 4,194,304  [B,128u,16kx,64]
  float* Y1im = Y1re + 4194304;

  k1_res   <<<4096, 256, 0, stream>>>(x, wlin, blin, res);
  k2_dftn  <<<4096, 256, 0, stream>>>(x, XAre, XAim);
  k3_dftm  <<< 512, 256, 0, stream>>>(XAre, XAim, XFre, XFim);
  k4_stage1<<<1024, 256, 0, stream>>>(XFre, XFim, ws0, ws1, C1re, C1im);
  k4_stage2<<<1024, 256, 0, stream>>>(C1re, C1im, ws0, ws1, Gre, Gim);
  k5_iffty <<< 512, 256, 0, stream>>>(Gre, Gim, Y1re, Y1im);
  k6_irfftx<<<4096, 256, 0, stream>>>(Y1re, Y1im, res, yact);
  k7_mlp   <<<4096, 256, 0, stream>>>(yact, wb1, bb1, wb2, bb2, wf1, bf1, wf2, bf2,
                                      out, out + 33554432);
}
